// WaveNetDecoder_5342939316254
// MI455X (gfx1250) — matrix operand order repaired and checked by reading, not compiled, not run on hardware
//
#include <hip/hip_runtime.h>

#define DEV static __device__ __forceinline__

typedef __bf16 v16bf __attribute__((ext_vector_type(16)));
typedef __bf16 v8bf  __attribute__((ext_vector_type(8)));
typedef float  v8f   __attribute__((ext_vector_type(8)));
typedef int    v4i   __attribute__((ext_vector_type(4)));

// address-space-qualified v4i for the async-to-LDS builtin (src: AS1, dst: AS3)
typedef __attribute__((address_space(1))) v4i gv4i;
typedef __attribute__((address_space(3))) v4i lv4i;

enum { Bn = 8, Cc = 128, Ll = 4096, Kt = 9, TN = 64, NT = TN / 16, KTI = Cc / 32 };

#if defined(__has_builtin)
#if __has_builtin(__builtin_amdgcn_global_load_async_to_lds_b128)
#define HAS_ASYNC_LDS 1
#endif
#endif
#ifndef HAS_ASYNC_LDS
#define HAS_ASYNC_LDS 0
#endif

// round-to-nearest-even fp32 -> bf16
DEV __bf16 f2bf(float f) {
  unsigned u = __builtin_bit_cast(unsigned, f);
  unsigned short s = (unsigned short)((u + 0x7FFFu + ((u >> 16) & 1u)) >> 16);
  return __builtin_bit_cast(__bf16, s);
}

#if HAS_ASYNC_LDS
// direct global -> LDS 16B async copy (ASYNCcnt tracked)
// builtin proto: (v4i AS1* src, v4i AS3* dst, imm offset, imm cpol)
DEV void async_copy_b128(const void* g, void* l) {
  gv4i* gp = (gv4i*)(unsigned long long)(size_t)g;        // 64-bit VA -> AS1
  lv4i* lp = (lv4i*)(unsigned int)(size_t)l;              // low 32 bits = LDS offset -> AS3
  __builtin_amdgcn_global_load_async_to_lds_b128(gp, lp, 0, 0);
}
DEV void wait_async() {
#if __has_builtin(__builtin_amdgcn_s_wait_asynccnt)
  __builtin_amdgcn_s_wait_asynccnt(0);
#else
  asm volatile("s_wait_asynccnt 0x0" ::: "memory");
#endif
}
#endif

// A fragment (16x32 bf16): lane<16: M=lane, K=kb+{0..15}, kb=kt*32
// lane>=16: kb += 16
DEV v16bf load_a(const __bf16* s, int mtile, int kt, int lane) {
  const int half = lane >> 4;
  const int m = mtile * 16 + (lane & 15);
  const int kb = kt * 32 + half * 16;
  const __bf16* row = s + m * Cc + kb;
  v8bf lo = *(const v8bf*)(row);
  v8bf hi = *(const v8bf*)(row + 8);
  v16bf a;
#pragma unroll
  for (int j = 0; j < 8; ++j) { a[j] = lo[j]; a[8 + j] = hi[j]; }
  return a;
}

// B fragment (32x16 bf16): lane -> N column, K packed 2/VGPR; sB is [time][channel]
// so the 16 halves are one contiguous 32B LDS read.
DEV v16bf load_b(const __bf16* s, int ntile, int kt, int lane) {
  const int half = lane >> 4;
  const int n = ntile * 16 + (lane & 15);
  const int kb = kt * 32 + half * 16;
  return *(const v16bf*)(s + n * Cc + kb);
}

// -------- weight pre-conversion: fp32 [l][O][I][K] -> bf16 tap-major [l][K][O][I] --------
__global__ void wn_prep(const float* __restrict__ Wf, const float* __restrict__ Wg,
                        const float* __restrict__ Wr,
                        __bf16* __restrict__ bWf, __bf16* __restrict__ bWg,
                        __bf16* __restrict__ bWr, int nConv, int nRes) {
  int j = blockIdx.x * blockDim.x + threadIdx.x;
  if (j < nConv) {
    int i = j & 127;
    int o = (j >> 7) & 127;
    int k = (j >> 14) % Kt;
    int l = j / (Kt * Cc * Cc);
    size_t src = ((size_t)(l * Cc + o) * Cc + i) * Kt + k;
    bWf[j] = f2bf(Wf[src]);
    bWg[j] = f2bf(Wg[src]);
  }
  if (j < nRes) bWr[j] = f2bf(Wr[j]); // [l][O][I] already contiguous
}

// ---------------- pre-conv: h = conv(x, Wp, pad=(1,1)) + bp (+ bf16 mirror) ----------------
// bf16 mirror is TIME-MAJOR [b][t][c] so layer tiles are row-contiguous for async copy.
__global__ __launch_bounds__(256)
void wn_pre(const float* __restrict__ x, float* __restrict__ hout,
            __bf16* __restrict__ hbout,
            const float* __restrict__ Wp, const float* __restrict__ bp) {
  __shared__ alignas(32) __bf16 sA[Cc * Cc];
  __shared__ alignas(32) __bf16 sB[TN * Cc];

  const int tid = threadIdx.x;
  const int lane = tid & 31;
  const int wv = tid >> 5;
  const int l0 = blockIdx.x * TN;
  const int batch = blockIdx.y;
  const size_t hbase = (size_t)batch * Cc * Ll;

  v8f acc[NT] = {};

#pragma unroll 1
  for (int k = 0; k < 3; ++k) {
    __syncthreads();
    for (int idx = tid; idx < TN * Cc; idx += 256) {
      int n = idx & (TN - 1);
      int c = idx >> 6;
      int t = l0 + n + k - 1;
      float v = (t >= 0 && t < Ll) ? x[hbase + (size_t)c * Ll + t] : 0.0f;
      sB[n * Cc + c] = f2bf(v);
    }
    for (int idx = tid; idx < Cc * Cc; idx += 256)
      sA[idx] = f2bf(Wp[(size_t)idx * 3 + k]);
    __syncthreads();
#pragma unroll
    for (int kt = 0; kt < KTI; ++kt) {
      v16bf a = load_a(sA, wv, kt, lane);
#pragma unroll
      for (int nt = 0; nt < NT; ++nt) {
        v16bf b = load_b(sB, nt, kt, lane);
        acc[nt] = __builtin_amdgcn_wmma_f32_16x16x32_bf16(
            false, a, false, b, (short)0, acc[nt], false, false);
      }
    }
  }

  const int half = lane >> 4;
  const int ncol = lane & 15;
#pragma unroll
  for (int nt = 0; nt < NT; ++nt) {
#pragma unroll
    for (int r = 0; r < 8; ++r) {
      int m = wv * 16 + half * 8 + r;
      int n = nt * 16 + ncol;
      float v = acc[nt][r] + bp[m];
      hout[hbase + (size_t)m * Ll + l0 + n] = v;
      if (hbout) hbout[((size_t)batch * Ll + l0 + n) * Cc + m] = f2bf(v);
    }
  }
}

// ------- one WaveNet layer: h' = Wr * (tanh(convF(h)+bf) * sig(convG(h)+bg)) + h -------
// fast=1: bf16 tap-major weights + time-major bf16 h mirror (async LDS fills)
__global__ __launch_bounds__(256)
void wn_layer(const float* __restrict__ hin, const __bf16* __restrict__ hbin,
              float* __restrict__ hout, __bf16* __restrict__ hbout,
              const float* __restrict__ Wf, const float* __restrict__ biasF,
              const float* __restrict__ Wg, const float* __restrict__ biasG,
              const float* __restrict__ Wr,
              const __bf16* __restrict__ bWf, const __bf16* __restrict__ bWg,
              const __bf16* __restrict__ bWr, int dil, int fast) {
  extern __shared__ __align__(32) char smem_raw[];
  __bf16* sAf = (__bf16*)smem_raw;             // 32 KB: Wf tap slice / Wr
  __bf16* sAg = sAf + Cc * Cc;                 // 32 KB: Wg tap slice
  __bf16* sB  = sAg + Cc * Cc;                 // 16 KB: shifted h tile [n][c]; later z

  const int tid = threadIdx.x;
  const int lane = tid & 31;
  const int wv = tid >> 5;
  const int l0 = blockIdx.x * TN;
  const int batch = blockIdx.y;
  const size_t hbase = (size_t)batch * Cc * Ll;

  v8f accF[NT] = {};
  v8f accG[NT] = {};

#pragma unroll 1
  for (int k = 0; k < Kt; ++k) {
    const int shift = (Kt - 1 - k) * dil; // causal: tap k reads h[l - (8-k)*d]
    __syncthreads();
    if (fast) {
#if HAS_ASYNC_LDS
      // activation tile rows are contiguous in the [t][c] mirror -> async B128 segments
      for (int idx = tid; idx < (TN * Cc) / 8; idx += 256) {
        int n = idx >> 4;
        int seg = idx & 15;
        int t = l0 + n - shift;
        __bf16* dst = sB + n * Cc + seg * 8;
        if (t >= 0) {
          async_copy_b128(hbin + ((size_t)batch * Ll + t) * Cc + seg * 8, dst);
        } else {
          uint4 z = {0, 0, 0, 0};
          *(uint4*)dst = z; // causal zero pad
        }
      }
      for (int idx = tid; idx < (Cc * Cc) / 8; idx += 256) {
        async_copy_b128(bWf + (size_t)k * Cc * Cc + (size_t)idx * 8, sAf + idx * 8);
        async_copy_b128(bWg + (size_t)k * Cc * Cc + (size_t)idx * 8, sAg + idx * 8);
      }
      wait_async();
#else
      for (int idx = tid; idx < TN * Cc; idx += 256) {
        int n = idx & (TN - 1);
        int c = idx >> 6;
        int t = l0 + n - shift;
        __bf16 zb = f2bf(0.0f);
        sB[n * Cc + c] = (t >= 0) ? hbin[((size_t)batch * Ll + t) * Cc + c] : zb;
      }
      const uint4* gf = (const uint4*)(bWf + (size_t)k * Cc * Cc);
      const uint4* gg = (const uint4*)(bWg + (size_t)k * Cc * Cc);
      uint4* lf = (uint4*)sAf;
      uint4* lg = (uint4*)sAg;
      for (int idx = tid; idx < Cc * Cc / 8; idx += 256) {
        lf[idx] = gf[idx];
        lg[idx] = gg[idx];
      }
#endif
    } else {
      for (int idx = tid; idx < TN * Cc; idx += 256) {
        int n = idx & (TN - 1);
        int c = idx >> 6;
        int t = l0 + n - shift;
        float v = (t >= 0) ? hin[hbase + (size_t)c * Ll + t] : 0.0f;
        sB[n * Cc + c] = f2bf(v);
      }
      for (int idx = tid; idx < Cc * Cc; idx += 256) {
        sAf[idx] = f2bf(Wf[(size_t)idx * Kt + k]);
        sAg[idx] = f2bf(Wg[(size_t)idx * Kt + k]);
      }
    }
    __syncthreads();
#pragma unroll
    for (int kt = 0; kt < KTI; ++kt) {
      v16bf af = load_a(sAf, wv, kt, lane);
      v16bf ag = load_a(sAg, wv, kt, lane);
#pragma unroll
      for (int nt = 0; nt < NT; ++nt) {
        v16bf b = load_b(sB, nt, kt, lane); // one B load feeds both WMMAs
        accF[nt] = __builtin_amdgcn_wmma_f32_16x16x32_bf16(
            false, af, false, b, (short)0, accF[nt], false, false);
        accG[nt] = __builtin_amdgcn_wmma_f32_16x16x32_bf16(
            false, ag, false, b, (short)0, accG[nt], false, false);
      }
    }
  }

  __syncthreads(); // all waves done reading sAf/sAg/sB
  const int half = lane >> 4;
  const int ncol = lane & 15;
  float bF[8], bG[8];
#pragma unroll
  for (int r = 0; r < 8; ++r) {
    int m = wv * 16 + half * 8 + r;
    bF[r] = biasF[m];
    bG[r] = biasG[m];
  }
  // gate -> z staged into sB as [n][c]
#pragma unroll
  for (int nt = 0; nt < NT; ++nt) {
#pragma unroll
    for (int r = 0; r < 8; ++r) {
      int m = wv * 16 + half * 8 + r;
      int n = nt * 16 + ncol;
      float fv = tanhf(accF[nt][r] + bF[r]);
      float gv = accG[nt][r] + bG[r];
      gv = 1.0f / (1.0f + __expf(-gv));
      sB[n * Cc + m] = f2bf(fv * gv);
    }
  }
  // Wr into sAf
  if (fast) {
#if HAS_ASYNC_LDS
    for (int idx = tid; idx < (Cc * Cc) / 8; idx += 256)
      async_copy_b128(bWr + (size_t)idx * 8, sAf + idx * 8);
    wait_async();
#else
    const uint4* gr = (const uint4*)bWr;
    uint4* lf = (uint4*)sAf;
    for (int idx = tid; idx < Cc * Cc / 8; idx += 256) lf[idx] = gr[idx];
#endif
  } else {
    for (int idx = tid; idx < Cc * Cc; idx += 256) sAf[idx] = f2bf(Wr[idx]);
  }
  __syncthreads();

  v8f accR[NT] = {};
#pragma unroll
  for (int kt = 0; kt < KTI; ++kt) {
    v16bf a = load_a(sAf, wv, kt, lane);
#pragma unroll
    for (int nt = 0; nt < NT; ++nt) {
      v16bf b = load_b(sB, nt, kt, lane);
      accR[nt] = __builtin_amdgcn_wmma_f32_16x16x32_bf16(
          false, a, false, b, (short)0, accR[nt], false, false);
    }
  }
#pragma unroll
  for (int nt = 0; nt < NT; ++nt) {
#pragma unroll
    for (int r = 0; r < 8; ++r) {
      int m = wv * 16 + half * 8 + r;
      int n = nt * 16 + ncol;
      size_t gi = hbase + (size_t)m * Ll + l0 + n;
      float v = accR[nt][r] + hin[gi]; // fp32 residual stream
      hout[gi] = v;
      if (hbout) hbout[((size_t)batch * Ll + l0 + n) * Cc + m] = f2bf(v);
    }
  }
}

__global__ void wn_add(float* __restrict__ out, const float* __restrict__ h, int n) {
  int i = blockIdx.x * blockDim.x + threadIdx.x;
  if (i < n) out[i] += h[i];
}

extern "C" void kernel_launch(void* const* d_in, const int* in_sizes, int n_in,
                              void* d_out, int out_size, void* d_ws, size_t ws_size,
                              hipStream_t stream) {
  const float* x    = (const float*)d_in[0];
  const float* preW = (const float*)d_in[1];
  const float* preB = (const float*)d_in[2];
  const float* Wf   = (const float*)d_in[3];
  const float* bf_  = (const float*)d_in[4];
  const float* Wg   = (const float*)d_in[5];
  const float* bg_  = (const float*)d_in[6];
  const float* Wr   = (const float*)d_in[7];

  const size_t nElem = (size_t)Bn * Cc * Ll;           // 4,194,304
  const size_t nConv = (size_t)20 * Kt * Cc * Cc;      // 2,949,120 per conv bank
  const size_t nRes  = (size_t)20 * Cc * Cc;           // 327,680

  char* ws = (char*)d_ws;
  size_t off = 0;
  float* h0 = (float*)(ws + off); off += nElem * 4;
  float* h1 = (float*)(ws + off); off += nElem * 4;
  const size_t baseNeed = off;
  __bf16* hb0 = (__bf16*)(ws + off); off += nElem * 2;
  __bf16* hb1 = (__bf16*)(ws + off); off += nElem * 2;
  __bf16* bWf = (__bf16*)(ws + off); off += nConv * 2;
  __bf16* bWg = (__bf16*)(ws + off); off += nConv * 2;
  __bf16* bWr = (__bf16*)(ws + off); off += nRes * 2;

  const int fast = (ws_size >= off && ws_size >= baseNeed) ? 1 : 0;

  dim3 grid(Ll / TN, Bn), block(256);
  const size_t smemBytes = (size_t)(2 * Cc * Cc + TN * Cc) * sizeof(__bf16); // 80 KB

  if (fast) {
    wn_prep<<<(int)((nConv + 255) / 256), 256, 0, stream>>>(
        Wf, Wg, Wr, bWf, bWg, bWr, (int)nConv, (int)nRes);
  }

  wn_pre<<<grid, block, 0, stream>>>(x, h0, fast ? hb0 : nullptr, preW, preB);

  float*  cur  = h0;  float*  nxt  = h1;
  __bf16* curb = hb0; __bf16* nxtb = hb1;
  for (int i = 0; i < 20; ++i) {
    int d = 1 << (i % 10);
    wn_layer<<<grid, block, smemBytes, stream>>>(
        cur, fast ? curb : nullptr, nxt, fast ? nxtb : nullptr,
        Wf + (size_t)i * Cc * Cc * Kt, bf_ + (size_t)i * Cc,
        Wg + (size_t)i * Cc * Cc * Kt, bg_ + (size_t)i * Cc,
        Wr + (size_t)i * Cc * Cc,
        fast ? bWf + (size_t)i * Kt * Cc * Cc : nullptr,
        fast ? bWg + (size_t)i * Kt * Cc * Cc : nullptr,
        fast ? bWr + (size_t)i * Cc * Cc : nullptr,
        d, fast);
    { float* t = cur; cur = nxt; nxt = t; }
    { __bf16* t = curb; curb = nxtb; nxtb = t; }
    if (i == 9)
      (void)hipMemcpyAsync(d_out, cur, nElem * sizeof(float),
                           hipMemcpyDeviceToDevice, stream);
  }
  // out = cycle1_out + h_final
  wn_add<<<(int)((nElem + 255) / 256), 256, 0, stream>>>((float*)d_out, cur, (int)nElem);
}
